// MultiHeadAttention_74612171866396
// MI455X (gfx1250) — compile-verified, matrix operand order checked by reading, not run on hardware
//
#include <hip/hip_runtime.h>

typedef __attribute__((ext_vector_type(16))) _Float16 v16h;
typedef __attribute__((ext_vector_type(8)))  _Float16 v8h;
typedef __attribute__((ext_vector_type(4)))  _Float16 v4h;
typedef __attribute__((ext_vector_type(8)))  float    v8f;

#define B_  4
#define S_  2048
#define D_  1024
#define H_  16
#define DK_ 64
#define M_  (B_ * S_)   // 8192 rows
#define BH_ (B_ * H_)   // 64 batch-heads

// ---------------------------------------------------------------------------
// WMMA helpers (CDNA5: D = A(16x32 f16) * B(32x16 f16) + C(16x16 f32))
// ---------------------------------------------------------------------------
__device__ __forceinline__ v8f wmma_f16(v16h a, v16h b, v8f c) {
  return __builtin_amdgcn_wmma_f32_16x16x32_f16(
      /*neg_a=*/false, a, /*neg_b=*/false, b,
      /*c_mod=*/(short)0, c, /*reuse_a=*/false, /*reuse_b=*/false);
}

// A-fragment (and column-major-B fragment) loader: 16x32 f16 tile, row-major.
// ISA 7.12.2: lane = {r = lane&15, half = lane>>4},
// elems j<8 -> K = half*8 + j ; j>=8 -> K = 16 + half*8 + (j-8).
__device__ __forceinline__ v16h load_frag_rm(const _Float16* p0, int ld) {
  const int lane = threadIdx.x & 31;
  const int r    = lane & 15;
  const int hf   = lane >> 4;
  const _Float16* p = p0 + (size_t)r * ld + hf * 8;
  v8h lo = *(const v8h*)(p);
  v8h hi = *(const v8h*)(p + 16);
  v16h f;
#pragma unroll
  for (int j = 0; j < 8; ++j) { f[j] = lo[j]; f[j + 8] = hi[j]; }
  return f;
}

// ---------------------------------------------------------------------------
// Kernel 1a: fp32 -> f16 bulk convert (vectorized x4)
// ---------------------------------------------------------------------------
__global__ __launch_bounds__(256) void cvt_f32_to_f16(
    const float* __restrict__ in, _Float16* __restrict__ out, int n4) {
  int i = blockIdx.x * 256 + threadIdx.x;
  if (i < n4) {
    float4 v = ((const float4*)in)[i];
    v4h o;
    o[0] = (_Float16)v.x; o[1] = (_Float16)v.y;
    o[2] = (_Float16)v.z; o[3] = (_Float16)v.w;
    ((v4h*)out)[i] = o;
  }
}

// ---------------------------------------------------------------------------
// Kernel 1b: W[k][n] fp32 -> WT[n][k] f16 (tiled transpose, 32x32)
// ---------------------------------------------------------------------------
__global__ __launch_bounds__(256) void transpose_to_f16(
    const float* __restrict__ W, _Float16* __restrict__ WT, int N, int K) {
  __shared__ float t[32][33];
  const int k0 = blockIdx.x * 32, n0 = blockIdx.y * 32;
  const int tx = threadIdx.x & 31, ty = threadIdx.x >> 5;  // ty in 0..7
  for (int i = ty; i < 32; i += 8)
    t[i][tx] = W[(size_t)(k0 + i) * N + (n0 + tx)];
  __syncthreads();
  for (int i = ty; i < 32; i += 8)
    WT[(size_t)(n0 + i) * K + (k0 + tx)] = (_Float16)t[tx][i];
}

// ---------------------------------------------------------------------------
// Kernel 2: projection GEMM.  For z in {0=q,1=k}: heads[z][bh][s][d]
//   = (X[z] @ Wq + bq) * scale (q pre-scaled by DK^-0.5).
// For z==2 (v): stored TRANSPOSED as vhT[bh][d][s] so the P@V B-fragments
//   become contiguous b128 loads; the store itself is one b128 per lane.
// WG = 256 thr = 8 waves (2x4); WG tile 64(M) x 128(N); wave tile 32x32.
// ---------------------------------------------------------------------------
__global__ __launch_bounds__(256) void proj_gemm_kernel(
    const _Float16* __restrict__ XH,   // [3][8192][1024] f16
    const _Float16* __restrict__ WQT,  // [1024][1024] f16, N-major
    const float* __restrict__ bq,      // [1024]
    _Float16* __restrict__ heads) {    // [3][64] x (2048*64) f16 slabs
  const int z = blockIdx.z;
  const _Float16* X = XH + (size_t)z * M_ * D_;
  const int wid  = threadIdx.x >> 5;
  const int wm   = wid >> 2;                 // 0..1
  const int wn   = wid & 3;                  // 0..3
  const int row0 = blockIdx.y * 64 + wm * 32;
  const int col0 = blockIdx.x * 128 + wn * 32;

  v8f acc[2][2] = {};
  for (int k = 0; k < D_; k += 32) {
    v16h a0 = load_frag_rm(X + (size_t)row0 * D_ + k, D_);
    v16h a1 = load_frag_rm(X + (size_t)(row0 + 16) * D_ + k, D_);
    v16h b0 = load_frag_rm(WQT + (size_t)col0 * D_ + k, D_);
    v16h b1 = load_frag_rm(WQT + (size_t)(col0 + 16) * D_ + k, D_);
    acc[0][0] = wmma_f16(a0, b0, acc[0][0]);
    acc[0][1] = wmma_f16(a0, b1, acc[0][1]);
    acc[1][0] = wmma_f16(a1, b0, acc[1][0]);
    acc[1][1] = wmma_f16(a1, b1, acc[1][1]);
  }

  const int lane = threadIdx.x & 31;
  const int ln = lane & 15, hf = lane >> 4;
  if (z == 2) {
    // vhT[bh][d][s]: lane's 8 C-rows are 8 consecutive s -> one b128 store
#pragma unroll
    for (int mt = 0; mt < 2; ++mt) {
#pragma unroll
      for (int nt = 0; nt < 2; ++nt) {
        const int col = col0 + nt * 16 + ln;
        const int hh = col >> 6, dd = col & 63;
        const float bias = bq[col];
        const int rbase = row0 + mt * 16 + 8 * hf;   // 8-aligned
        const int bb = rbase >> 11, ss = rbase & 2047;
        v8h o;
#pragma unroll
        for (int r = 0; r < 8; ++r) o[r] = (_Float16)(acc[mt][nt][r] + bias);
        *(v8h*)(heads + (size_t)(2 * BH_ + bb * H_ + hh) * (S_ * DK_) +
                (size_t)dd * S_ + ss) = o;
      }
    }
  } else {
    const float scale = (z == 0) ? 0.125f : 1.0f;  // DK^-0.5 onto qh
#pragma unroll
    for (int mt = 0; mt < 2; ++mt) {
#pragma unroll
      for (int nt = 0; nt < 2; ++nt) {
        const int col  = col0 + nt * 16 + ln;
        const int hh   = col >> 6, dd = col & 63;
        const float bias = bq[col];
#pragma unroll
        for (int r = 0; r < 8; ++r) {
          const int row = row0 + mt * 16 + r + 8 * hf;
          const int bb = row >> 11, ss = row & 2047;
          heads[(size_t)(z * BH_ + bb * H_ + hh) * (S_ * DK_) +
                (size_t)ss * DK_ + dd] =
              (_Float16)((acc[mt][nt][r] + bias) * scale);
        }
      }
    }
  }
}

// ---------------------------------------------------------------------------
// Kernel 3: fused attention for one (bh, 16 query rows).
// Phase 1: scores 16x2048 via WMMA -> LDS f16 (64 KB)
// Phase 2a: per-row max / sum(exp) via 16-lane shuffles -> rowinv[16]
// Phase 2b: block-flat normalize: coalesced float4 writes of attn to HBM
//           (the one mandatory 1 GiB stream), f16 probs kept in LDS
// Phase 3: P @ V via WMMA, all 8 waves (4 n-tiles x 2 K-halves); partials
//          reduced through the (now dead) score LDS -> ctx f16
// ---------------------------------------------------------------------------
__global__ __launch_bounds__(256) void attn_fused_kernel(
    const _Float16* __restrict__ heads,  // slabs as produced above
    float* __restrict__ attn_out,        // [64][2048][2048] f32
    _Float16* __restrict__ ctx) {        // [8192][1024] f16
  __shared__ __align__(16) _Float16 sc[16 * S_];  // 65536 bytes
  __shared__ float rowinv[16];

  const int bh = blockIdx.y;
  const int q0 = blockIdx.x * 16;
  const _Float16* qh  = heads + ((size_t)(0 * BH_ + bh) * S_ + q0) * DK_;
  const _Float16* kh  = heads + (size_t)(1 * BH_ + bh) * (S_ * DK_);
  const _Float16* vhT = heads + (size_t)(2 * BH_ + bh) * (S_ * DK_);

  const int wid  = threadIdx.x >> 5;
  const int lane = threadIdx.x & 31;
  const int ln = lane & 15, hf = lane >> 4;

  // ---- Phase 1: S = (qh*scale) @ kh^T ; each wave owns 16 of 128 n-tiles
  {
    v16h a0 = load_frag_rm(qh + 0, DK_);
    v16h a1 = load_frag_rm(qh + 32, DK_);
    for (int nt = wid * 16; nt < wid * 16 + 16; ++nt) {
      v16h b0 = load_frag_rm(kh + (size_t)nt * 16 * DK_ + 0, DK_);
      v16h b1 = load_frag_rm(kh + (size_t)nt * 16 * DK_ + 32, DK_);
      v8f c = {};
      c = wmma_f16(a0, b0, c);
      c = wmma_f16(a1, b1, c);
#pragma unroll
      for (int r = 0; r < 8; ++r)
        sc[(size_t)(r + 8 * hf) * S_ + nt * 16 + ln] = (_Float16)c[r];
    }
  }
  __syncthreads();

  // ---- Phase 2a: row stats; 16 lanes per row, stride-16 columns
  {
    const int row = threadIdx.x >> 4;  // 0..15
    const int tc  = threadIdx.x & 15;  // 0..15
    float m = -3.0e38f;
    for (int c0 = tc; c0 < S_; c0 += 16)
      m = fmaxf(m, (float)sc[(size_t)row * S_ + c0]);
#pragma unroll
    for (int i = 1; i < 16; i <<= 1) m = fmaxf(m, __shfl_xor(m, i, 16));
    float ssum = 0.0f;
    for (int c0 = tc; c0 < S_; c0 += 16) {
      float e = __expf((float)sc[(size_t)row * S_ + c0] - m);
      ssum += e;
      sc[(size_t)row * S_ + c0] = (_Float16)e;   // unnormalized exp
    }
#pragma unroll
    for (int i = 1; i < 16; i <<= 1) ssum += __shfl_xor(ssum, i, 16);
    if (tc == 0) rowinv[row] = 1.0f / ssum;
  }
  __syncthreads();

  // ---- Phase 2b: normalize + fully-coalesced float4 attn write
  {
    float* abase = attn_out + ((size_t)bh * S_ + q0) * S_;
#pragma unroll 4
    for (int i = 0; i < 32; ++i) {
      const int c   = threadIdx.x + 256 * i;   // 0..8191 chunk id
      const int rw  = c >> 9;                  // row 0..15
      const int col = (c & 511) * 4;           // col 0..2044
      v4h pv = *(v4h*)(sc + (size_t)rw * S_ + col);
      const float inv = rowinv[rw];
      float4 o;
      o.x = (float)pv[0] * inv;
      o.y = (float)pv[1] * inv;
      o.z = (float)pv[2] * inv;
      o.w = (float)pv[3] * inv;
      pv[0] = (_Float16)o.x; pv[1] = (_Float16)o.y;
      pv[2] = (_Float16)o.z; pv[3] = (_Float16)o.w;
      *(v4h*)(sc + (size_t)rw * S_ + col) = pv;          // probs for P@V
      *(float4*)(abase + (size_t)rw * S_ + col) = o;     // coalesced HBM write
    }
  }
  __syncthreads();

  // ---- Phase 3: out[16x64] = P[16x2048] @ V[2048x64]; 8 waves =
  //      4 n-tiles x 2 K-halves; B from vhT is contiguous b128 loads.
  {
    const int ntile = wid & 3;
    const int khalf = wid >> 2;
    v8f acc = {};
    for (int kt = khalf * 32; kt < khalf * 32 + 32; ++kt) {
      v16h a = load_frag_rm(sc + kt * 32, S_);                       // LDS probs
      v16h b = load_frag_rm(vhT + (size_t)(ntile * 16) * S_ + kt * 32, S_);
      acc = wmma_f16(a, b, acc);
    }
    __syncthreads();                     // all probs reads done; sc reusable
    float* pbuf = (float*)sc;            // 4 KB partial-sum buffer
    if (khalf == 1) {
#pragma unroll
      for (int r = 0; r < 8; ++r)
        pbuf[ntile * 256 + (r + 8 * hf) * 16 + ln] = acc[r];
    }
    __syncthreads();
    if (khalf == 0) {
      const int bb = bh >> 4, hh = bh & 15;
#pragma unroll
      for (int r = 0; r < 8; ++r) {
        const float v2 = acc[r] + pbuf[ntile * 256 + (r + 8 * hf) * 16 + ln];
        ctx[(size_t)(bb * S_ + q0 + r + 8 * hf) * D_ + hh * DK_ +
            ntile * 16 + ln] = (_Float16)v2;
      }
    }
  }
}

// ---------------------------------------------------------------------------
// Kernel 4: FC GEMM  Y = ctx @ Wfc + bfc + residual   (fp32 out)
// ---------------------------------------------------------------------------
__global__ __launch_bounds__(256) void fc_gemm_kernel(
    const _Float16* __restrict__ CTX,   // [8192][1024] f16
    const _Float16* __restrict__ WFCT,  // [1024][1024] f16, N-major
    const float* __restrict__ bfc,      // [1024]
    const float* __restrict__ resid,    // q, [8192][1024] f32
    float* __restrict__ Y) {            // [8192][1024] f32
  const int wid  = threadIdx.x >> 5;
  const int wm   = wid >> 2;
  const int wn   = wid & 3;
  const int row0 = blockIdx.y * 64 + wm * 32;
  const int col0 = blockIdx.x * 128 + wn * 32;

  v8f acc[2][2] = {};
  for (int k = 0; k < D_; k += 32) {
    v16h a0 = load_frag_rm(CTX + (size_t)row0 * D_ + k, D_);
    v16h a1 = load_frag_rm(CTX + (size_t)(row0 + 16) * D_ + k, D_);
    v16h b0 = load_frag_rm(WFCT + (size_t)col0 * D_ + k, D_);
    v16h b1 = load_frag_rm(WFCT + (size_t)(col0 + 16) * D_ + k, D_);
    acc[0][0] = wmma_f16(a0, b0, acc[0][0]);
    acc[0][1] = wmma_f16(a0, b1, acc[0][1]);
    acc[1][0] = wmma_f16(a1, b0, acc[1][0]);
    acc[1][1] = wmma_f16(a1, b1, acc[1][1]);
  }

  const int lane = threadIdx.x & 31;
  const int ln = lane & 15, hf = lane >> 4;
#pragma unroll
  for (int mt = 0; mt < 2; ++mt) {
#pragma unroll
    for (int nt = 0; nt < 2; ++nt) {
      const int col = col0 + nt * 16 + ln;
      const float bias = bfc[col];
#pragma unroll
      for (int r = 0; r < 8; ++r) {
        const int row = row0 + mt * 16 + r + 8 * hf;
        const size_t idx = (size_t)row * D_ + col;
        Y[idx] = acc[mt][nt][r] + bias + resid[idx];
      }
    }
  }
}

// ---------------------------------------------------------------------------
// Kernel 5: LayerNorm over D=1024 per row -> d_out x region
// ---------------------------------------------------------------------------
__global__ __launch_bounds__(256) void layernorm_kernel(
    const float* __restrict__ Y, const float* __restrict__ gamma,
    const float* __restrict__ beta, float* __restrict__ out) {
  __shared__ float red[16];
  const int row = blockIdx.x;
  const int c = threadIdx.x * 4;
  const float4 v = *(const float4*)(Y + (size_t)row * D_ + c);
  float s  = v.x + v.y + v.z + v.w;
  float s2 = v.x * v.x + v.y * v.y + v.z * v.z + v.w * v.w;
#pragma unroll
  for (int i = 1; i < 32; i <<= 1) {
    s  += __shfl_xor(s, i, 32);
    s2 += __shfl_xor(s2, i, 32);
  }
  if ((threadIdx.x & 31) == 0) {
    red[(threadIdx.x >> 5) * 2]     = s;
    red[(threadIdx.x >> 5) * 2 + 1] = s2;
  }
  __syncthreads();
  float ts = 0.f, ts2 = 0.f;
#pragma unroll
  for (int w = 0; w < 8; ++w) { ts += red[w * 2]; ts2 += red[w * 2 + 1]; }
  const float mu  = ts * (1.0f / D_);
  const float var = ts2 * (1.0f / D_) - mu * mu;
  const float rs  = rsqrtf(var + 1e-6f);
  const float4 g  = *(const float4*)(gamma + c);
  const float4 bt = *(const float4*)(beta + c);
  float4 o;
  o.x = (v.x - mu) * rs * g.x + bt.x;
  o.y = (v.y - mu) * rs * g.y + bt.y;
  o.z = (v.z - mu) * rs * g.z + bt.z;
  o.w = (v.w - mu) * rs * g.w + bt.w;
  *(float4*)(out + (size_t)row * D_ + c) = o;
}

// ---------------------------------------------------------------------------
// Host orchestration
// ---------------------------------------------------------------------------
extern "C" void kernel_launch(void* const* d_in, const int* in_sizes, int n_in,
                              void* d_out, int out_size, void* d_ws,
                              size_t ws_size, hipStream_t stream) {
  (void)in_sizes; (void)n_in; (void)out_size; (void)ws_size;
  const float* q     = (const float*)d_in[0];
  const float* k     = (const float*)d_in[1];
  const float* v     = (const float*)d_in[2];
  const float* Wq    = (const float*)d_in[3];
  const float* bq    = (const float*)d_in[4];
  const float* Wfc   = (const float*)d_in[5];
  const float* bfc   = (const float*)d_in[6];
  const float* gamma = (const float*)d_in[7];
  const float* beta  = (const float*)d_in[8];

  float* out_x    = (float*)d_out;                       // [4,2048,1024]
  float* out_attn = out_x + (size_t)M_ * D_;             // [4,16,2048,2048]

  // Workspace layout (bytes). Y (fp32, 32 MiB) aliases XH (dead by then).
  char* ws = (char*)d_ws;
  constexpr size_t OFF_XH    = 0;                        // 48 MiB
  constexpr size_t OFF_WQT   = OFF_XH   + (size_t)3 * M_ * D_ * 2;
  constexpr size_t OFF_WFCT  = OFF_WQT  + (size_t)D_ * D_ * 2;
  constexpr size_t OFF_HEADS = OFF_WFCT + (size_t)D_ * D_ * 2;   // 48 MiB
  constexpr size_t OFF_CTX   = OFF_HEADS + (size_t)3 * BH_ * S_ * DK_ * 2;
  _Float16* XH    = (_Float16*)(ws + OFF_XH);
  _Float16* WQT   = (_Float16*)(ws + OFF_WQT);
  _Float16* WFCT  = (_Float16*)(ws + OFF_WFCT);
  _Float16* HEADS = (_Float16*)(ws + OFF_HEADS);
  _Float16* CTX   = (_Float16*)(ws + OFF_CTX);
  float*    Y     = (float*)(ws + OFF_XH);               // alias (sequential)

  const int n4 = (M_ * D_) / 4;  // 2097152
  cvt_f32_to_f16<<<dim3(n4 / 256), 256, 0, stream>>>(q, XH + 0 * (size_t)M_ * D_, n4);
  cvt_f32_to_f16<<<dim3(n4 / 256), 256, 0, stream>>>(k, XH + 1 * (size_t)M_ * D_, n4);
  cvt_f32_to_f16<<<dim3(n4 / 256), 256, 0, stream>>>(v, XH + 2 * (size_t)M_ * D_, n4);
  transpose_to_f16<<<dim3(D_ / 32, D_ / 32), 256, 0, stream>>>(Wq,  WQT,  D_, D_);
  transpose_to_f16<<<dim3(D_ / 32, D_ / 32), 256, 0, stream>>>(Wfc, WFCT, D_, D_);

  proj_gemm_kernel<<<dim3(D_ / 128, M_ / 64, 3), 256, 0, stream>>>(XH, WQT, bq, HEADS);

  attn_fused_kernel<<<dim3(S_ / 16, BH_), 256, 0, stream>>>(HEADS, out_attn, CTX);

  fc_gemm_kernel<<<dim3(D_ / 128, M_ / 64), 256, 0, stream>>>(CTX, WFCT, bfc, q, Y);

  layernorm_kernel<<<dim3(M_), 256, 0, stream>>>(Y, gamma, beta, out_x);
}
